// SpectralLocalRefiner_9234179686570
// MI455X (gfx1250) — compile-verified, matrix operand order checked by reading, not run on hardware
//
#include <hip/hip_runtime.h>
#include <math.h>

// ---------------- problem constants ----------------
#define B_    2
#define S_    8192
#define DM    1024
#define NH    16
#define HD    64
#define WIN   512
#define KSEL  1024
#define CHUNK_ 256
#define RMSEPS 1.1920929e-07f

typedef __bf16 v16bf __attribute__((ext_vector_type(16)));
typedef float  v8f   __attribute__((ext_vector_type(8)));
typedef int    v4i   __attribute__((ext_vector_type(4)));

__device__ __forceinline__ int kpat(int i, int hf) {
  // ISA 16-bit A/B layout: VGPR i holds K pair {kk, kk+1}; lanes>=16 get +8
  return ((i < 4) ? (2 * i) : (16 + 2 * (i - 4))) + 8 * hf;
}
__device__ __forceinline__ v8f vzero8() {
  v8f v; for (int i = 0; i < 8; ++i) v[i] = 0.f; return v;
}
__device__ __forceinline__ unsigned short bfbits(float f) {
  return __builtin_bit_cast(unsigned short, (__bf16)f);  // native cvt, RNE
}

// ---------------- CDNA5 async global->LDS copy (ASYNCcnt) ----------------
#if __has_builtin(__builtin_amdgcn_global_load_async_to_lds_b128)
#define HAVE_ASYNC_LDS 1
__device__ __forceinline__ void async_copy16(void* lds_dst, const void* g_src) {
  __builtin_amdgcn_global_load_async_to_lds_b128(
      (__attribute__((address_space(1))) v4i*)g_src,
      (__attribute__((address_space(3))) v4i*)lds_dst, 0, 0);
}
template <int N> __device__ __forceinline__ void wait_async() {
#if __has_builtin(__builtin_amdgcn_s_wait_asynccnt)
  __builtin_amdgcn_s_wait_asynccnt(N);
#else
  asm volatile("s_wait_asynccnt %0" ::"n"(N) : "memory");
#endif
}
#else
#define HAVE_ASYNC_LDS 0
__device__ __forceinline__ void async_copy16(void* lds_dst, const void* g_src) {
  *(uint4*)lds_dst = *(const uint4*)g_src;  // synchronous fallback
}
template <int N> __device__ __forceinline__ void wait_async() {}
#endif

// ---------------- 0: zero output ----------------
__global__ __launch_bounds__(256) void zero_kernel(float4* p, int n4) {
  int i = blockIdx.x * 256 + threadIdx.x;
  if (i < n4) p[i] = make_float4(0.f, 0.f, 0.f, 0.f);
}

// ---------------- 1: routing scores ----------------
__global__ __launch_bounds__(256) void score_kernel(const float* __restrict__ x,
                                                    const float* __restrict__ ch,
                                                    float* __restrict__ scores) {
  int wave = threadIdx.x >> 5, lane = threadIdx.x & 31;
  long long row = (long long)blockIdx.x * 8 + wave;
  if (row >= (long long)B_ * (S_ - 1)) return;
  int b = (int)(row / (S_ - 1)), s = (int)(row % (S_ - 1));
  const float* xp = x + ((long long)b * S_ + s + 1) * DM;
  const float* cp = ch + ((long long)b * S_ + s) * DM;
  float acc = 0.f;
  for (int d = lane; d < DM; d += 32) {
    float xv = xp[d];
    float ax = fmaxf(fabsf(xv), 1e-7f);
    float mag = 1.f - 1.f / (1.f + ax);
    float ss = (xv > 0.f) ? mag : (xv < 0.f ? -mag : 0.f);
    acc += fabsf(ss - cp[d]);
  }
  for (int o = 1; o < 32; o <<= 1) acc += __shfl_xor(acc, o, 32);
  if (lane == 0) scores[(long long)b * S_ + s] = acc * (1.f / DM);
}

// ---------------- 2: last score = max ----------------
__global__ __launch_bounds__(256) void finalize_max_kernel(float* __restrict__ scores) {
  __shared__ float red[256];
  int b = blockIdx.x, t = threadIdx.x;
  float m = -1e30f;
  for (int s = t; s < S_ - 1; s += 256) m = fmaxf(m, scores[(long long)b * S_ + s]);
  red[t] = m; __syncthreads();
  for (int o = 128; o > 0; o >>= 1) { if (t < o) red[t] = fmaxf(red[t], red[t + o]); __syncthreads(); }
  if (t == 0) scores[(long long)b * S_ + (S_ - 1)] = red[0];
}

// ---------------- 3: per-batch top-K via bitonic sort ----------------
__global__ __launch_bounds__(1024) void topk_kernel(const float* __restrict__ scores,
                                                    int* __restrict__ idxOut) {
  __shared__ float sk[S_];
  __shared__ int   si[S_];
  int b = blockIdx.x, t = threadIdx.x;
  for (int j = t; j < S_; j += 1024) { sk[j] = scores[(long long)b * S_ + j]; si[j] = j; }
  __syncthreads();
  for (int k = 2; k <= S_; k <<= 1) {             // sort (score desc, idx asc)
    for (int j = k >> 1; j > 0; j >>= 1) {
      for (int i = t; i < S_; i += 1024) {
        int l = i ^ j;
        if (l > i) {
          float a = sk[i], c = sk[l];
          int ai = si[i], ci = si[l];
          bool iBetter = (a > c) || (a == c && ai < ci);
          bool dirDesc = ((i & k) == 0);
          bool sw = dirDesc ? !iBetter : iBetter;
          if (sw) { sk[i] = c; sk[l] = a; si[i] = ci; si[l] = ai; }
        }
      }
      __syncthreads();
    }
  }
  for (int k = 2; k <= KSEL; k <<= 1) {           // sort top-K indices ascending
    for (int j = k >> 1; j > 0; j >>= 1) {
      int i = t, l = i ^ j;
      if (t < KSEL && l > i) {
        int a = si[i], c = si[l];
        bool asc = ((i & k) == 0);
        bool sw = asc ? (a > c) : (a < c);
        if (sw) { si[i] = c; si[l] = a; }
      }
      __syncthreads();
    }
  }
  if (t < KSEL) idxOut[b * KSEL + t] = si[t];
}

// ---------------- 4: f32 -> bf16 bulk convert ----------------
__global__ __launch_bounds__(256) void cvt_bf16_kernel(const float4* __restrict__ in,
                                                       ushort4* __restrict__ out, int n4) {
  int t = blockIdx.x * 256 + threadIdx.x;
  if (t >= n4) return;
  float4 v = in[t];
  ushort4 o;
  o.x = bfbits(v.x); o.y = bfbits(v.y); o.z = bfbits(v.z); o.w = bfbits(v.w);
  out[t] = o;
}

// ---------------- 5: gather selected rows of x, convert to bf16 ----------------
__global__ __launch_bounds__(256) void gather_cvt_kernel(const float* __restrict__ x,
                                                         const int* __restrict__ idx,
                                                         ushort4* __restrict__ xsel) {
  int t = blockIdx.x * 256 + threadIdx.x;   // 2^19 threads total
  int c4 = t & 255;                         // 256 float4 per row
  int m = (t >> 8) & 1023;
  int b = t >> 18;
  int srow = idx[b * KSEL + m];
  float4 v = ((const float4*)(x + ((long long)b * S_ + srow) * DM))[c4];
  ushort4 o;
  o.x = bfbits(v.x); o.y = bfbits(v.y); o.z = bfbits(v.z); o.w = bfbits(v.w);
  xsel[((long long)b * KSEL + m) * (DM / 4) + c4] = o;
}

// ---------------- 6: bf16 WMMA GEMM  C(M,N) = A(M,D) * W(N,D)^T ----------------
// Double-buffered async-to-LDS tile staging (ASYNCcnt pipelined).
__global__ __launch_bounds__(256) void gemm_bf16(const __bf16* __restrict__ A,
                                                 long long aBatch, int lda,
                                                 const __bf16* __restrict__ W,
                                                 float* __restrict__ out,
                                                 long long oBatch, int ldo,
                                                 int N, int D) {
  __shared__ __bf16 At[2][128][40];   // row stride 80B (16B-aligned, conflict-free)
  __shared__ __bf16 Wt[2][128][40];
  int b = blockIdx.z;
  int m0 = blockIdx.x * 128, n0 = blockIdx.y * 128;
  int tid = threadIdx.x, wave = tid >> 5, lane = tid & 31;
  int hf = lane >> 4, l16 = lane & 15;
  int wr = wave & 1, wc = wave >> 1;
  v8f acc[4][2];
  for (int a = 0; a < 4; ++a) for (int j = 0; j < 2; ++j) acc[a][j] = vzero8();

  // staging coords: thread -> (row, 2 of 4 16B segments)
  int r = tid >> 1;
  int sp = (tid & 1) * 2;
  const __bf16* aRow = A + (long long)b * aBatch + (long long)(m0 + r) * lda;
  const __bf16* wRow = W + (long long)(n0 + r) * D;

  int nsteps = D / 32;
#define ISSUE_TILE(buf, d0)                                            \
  do {                                                                 \
    async_copy16(&At[buf][r][sp * 8], aRow + (d0) + sp * 8);           \
    async_copy16(&At[buf][r][(sp + 1) * 8], aRow + (d0) + (sp + 1) * 8); \
    async_copy16(&Wt[buf][r][sp * 8], wRow + (d0) + sp * 8);           \
    async_copy16(&Wt[buf][r][(sp + 1) * 8], wRow + (d0) + (sp + 1) * 8); \
  } while (0)

  ISSUE_TILE(0, 0);
  for (int ks = 0; ks < nsteps; ++ks) {
    int cur = ks & 1;
    if (ks + 1 < nsteps) { ISSUE_TILE(cur ^ 1, (ks + 1) * 32); wait_async<4>(); }
    else { wait_async<0>(); }
    __syncthreads();
    v16bf bf[2];
    for (int j = 0; j < 2; ++j) {
      int nc = wc * 32 + j * 16 + l16;
      for (int i = 0; i < 8; ++i) {
        int kk = kpat(i, hf);
        bf[j][2 * i] = Wt[cur][nc][kk];
        bf[j][2 * i + 1] = Wt[cur][nc][kk + 1];
      }
    }
    for (int qi = 0; qi < 4; ++qi) {
      int mr = wr * 64 + qi * 16 + l16;
      v16bf af;
      for (int i = 0; i < 8; ++i) {
        int kk = kpat(i, hf);
        af[2 * i] = At[cur][mr][kk];
        af[2 * i + 1] = At[cur][mr][kk + 1];
      }
      for (int j = 0; j < 2; ++j)
        acc[qi][j] = __builtin_amdgcn_wmma_f32_16x16x32_bf16(
            false, af, false, bf[j], (short)0, acc[qi][j], false, false);
    }
    __syncthreads();
  }
#undef ISSUE_TILE
  float* ob = out + (long long)b * oBatch;
  for (int qi = 0; qi < 4; ++qi)
    for (int j = 0; j < 2; ++j)
      for (int rr2 = 0; rr2 < 8; ++rr2) {
        int mrow = m0 + wr * 64 + qi * 16 + rr2 + 8 * hf;
        int ncol = n0 + wc * 32 + j * 16 + l16;
        ob[(long long)mrow * ldo + ncol] = acc[qi][j][rr2];
      }
}

// ---------------- 7: RoPE + pack q1/q2/k/v to per-head bf16 ----------------
__global__ __launch_bounds__(256) void rope_pack_kernel(const float* __restrict__ qkv,
                                                        const int* __restrict__ idx,
                                                        __bf16* __restrict__ qp,
                                                        __bf16* __restrict__ kp,
                                                        __bf16* __restrict__ vp) {
  int t = blockIdx.x * 256 + threadIdx.x;   // 2^20 threads: (b, kr, h, i)
  int i = t & 31;
  int h = (t >> 5) & 15;
  int kr = (t >> 9) & 1023;
  int b = t >> 19;
  int pos = idx[b * KSEL + kr];
  float freq = (float)pos * __expf(-(float)i * 0.28782313662425572f);  // ln(1e4)/32
  float c = __cosf(freq), sn = __sinf(freq);
  long long base = ((long long)b * KSEL + kr) * 4096 + h * HD + 2 * i;
  for (int st = 0; st < 2; ++st) {
    float x1 = qkv[base + st * DM], x2 = qkv[base + st * DM + 1];
    long long o = (((long long)(b * NH + h) * 2 + st) * KSEL + kr) * HD + 2 * i;
    qp[o] = (__bf16)(x1 * c - x2 * sn);
    qp[o + 1] = (__bf16)(x1 * sn + x2 * c);
  }
  {
    float x1 = qkv[base + 2 * DM], x2 = qkv[base + 2 * DM + 1];
    long long o = ((long long)(b * NH + h) * KSEL + kr) * HD + 2 * i;
    kp[o] = (__bf16)(x1 * c - x2 * sn);
    kp[o + 1] = (__bf16)(x1 * sn + x2 * c);
    vp[o] = (__bf16)qkv[base + 3 * DM];
    vp[o + 1] = (__bf16)qkv[base + 3 * DM + 1];
  }
}

// ---------------- 8: differential local attention (flash-style, bf16 WMMA) ----------------
__global__ __launch_bounds__(256) void attn_kernel(const __bf16* __restrict__ qp,
                                                   const __bf16* __restrict__ kp,
                                                   const __bf16* __restrict__ vp,
                                                   const int* __restrict__ gidx,
                                                   const float* __restrict__ lambda_logit,
                                                   float* __restrict__ diffOut) {
  __shared__ __bf16 kt[256][72];   // row stride 144B (16B-aligned, conflict-free)
  __shared__ __bf16 vt[256][72];
  __shared__ int posL[512];
  __shared__ __bf16 pscr[8][4][16][32];

  const int qc = blockIdx.x, h = blockIdx.y, b = blockIdx.z;
  const int tid = threadIdx.x, wave = tid >> 5, lane = tid & 31;
  const int hf = lane >> 4, l16 = lane & 15;
  const int q_start = qc * CHUNK_;
  const int k_start = (q_start >= CHUNK_) ? (q_start - CHUNK_) : 0;
  const int nk = q_start + CHUNK_ - k_start;   // 256 or 512

  for (int j = tid; j < nk; j += 256) posL[j] = gidx[b * KSEL + k_start + j];
  __syncthreads();

  // Q fragments from packed bf16: [qsub][stream][dhalf]
  v16bf qf[2][2][2];
  for (int qsub = 0; qsub < 2; ++qsub) {
    int mrow = q_start + wave * 32 + qsub * 16 + l16;
    for (int st = 0; st < 2; ++st) {
      const __bf16* src = qp + (((long long)(b * NH + h) * 2 + st) * KSEL + mrow) * HD;
      for (int dh = 0; dh < 2; ++dh)
        for (int i = 0; i < 8; ++i) {
          int kk = kpat(i, hf);
          qf[qsub][st][dh][2 * i] = src[dh * 32 + kk];
          qf[qsub][st][dh][2 * i + 1] = src[dh * 32 + kk + 1];
        }
    }
  }
  int pq[2][8];
  for (int qsub = 0; qsub < 2; ++qsub)
    for (int r = 0; r < 8; ++r)
      pq[qsub][r] = posL[q_start - k_start + wave * 32 + qsub * 16 + r + 8 * hf];
  const int pqmin = posL[q_start - k_start];

  float mrw[2][2][8], lrw[2][2][8];
  v8f oacc[2][2][4];
  for (int a = 0; a < 2; ++a)
    for (int s = 0; s < 2; ++s) {
      for (int r = 0; r < 8; ++r) { mrw[a][s][r] = -1e30f; lrw[a][s][r] = 0.f; }
      for (int n = 0; n < 4; ++n) oacc[a][s][n] = vzero8();
    }
  const float scale = 0.125f;  // 1/sqrt(64)
  const __bf16* kBase = kp + ((long long)(b * NH + h) * KSEL + k_start) * HD;
  const __bf16* vBase = vp + ((long long)(b * NH + h) * KSEL + k_start) * HD;

  for (int s0 = 0; s0 < nk; s0 += 256) {
    // async-stage K/V rows [s0, s0+256) into LDS (raw bf16, no VALU)
    {
      long long roff = (long long)(s0 + tid) * HD;
      for (int s = 0; s < 4; ++s) {
        async_copy16(&kt[tid][s * 8], kBase + roff + s * 8);
        async_copy16(&vt[tid][s * 8], vBase + roff + s * 8);
      }
      wait_async<0>();
    }
    __syncthreads();

    for (int kb = 0; kb < 256; kb += 32) {
      int kg = s0 + kb;
      if (posL[kg + 31] + WIN <= pqmin) continue;  // uniform out-of-window skip
      int pk0 = posL[kg + l16];
      int pk1 = posL[kg + 16 + l16];

      v8f sc[2][2][2];
      for (int a = 0; a < 2; ++a) for (int s = 0; s < 2; ++s) for (int j = 0; j < 2; ++j)
        sc[a][s][j] = vzero8();
      for (int dh = 0; dh < 2; ++dh) {
        v16bf bk0, bk1;
        for (int i = 0; i < 8; ++i) {
          int kk = kpat(i, hf);
          bk0[2 * i] = kt[kb + l16][dh * 32 + kk];
          bk0[2 * i + 1] = kt[kb + l16][dh * 32 + kk + 1];
          bk1[2 * i] = kt[kb + 16 + l16][dh * 32 + kk];
          bk1[2 * i + 1] = kt[kb + 16 + l16][dh * 32 + kk + 1];
        }
        for (int qsub = 0; qsub < 2; ++qsub)
          for (int st = 0; st < 2; ++st) {
            sc[qsub][st][0] = __builtin_amdgcn_wmma_f32_16x16x32_bf16(
                false, qf[qsub][st][dh], false, bk0, (short)0, sc[qsub][st][0], false, false);
            sc[qsub][st][1] = __builtin_amdgcn_wmma_f32_16x16x32_bf16(
                false, qf[qsub][st][dh], false, bk1, (short)0, sc[qsub][st][1], false, false);
          }
      }

      // mask + online softmax; P -> per-wave LDS scratch (A-layout round trip)
      for (int qsub = 0; qsub < 2; ++qsub)
        for (int st = 0; st < 2; ++st) {
          int combo = qsub * 2 + st;
          for (int r = 0; r < 8; ++r) {
            float v0 = sc[qsub][st][0][r] * scale;
            float v1 = sc[qsub][st][1][r] * scale;
            int p = pq[qsub][r];
            if (!((pk0 <= p) && (p - pk0 < WIN))) v0 = -1e9f;
            if (!((pk1 <= p) && (p - pk1 < WIN))) v1 = -1e9f;
            float t = fmaxf(v0, v1);
            for (int o = 1; o < 16; o <<= 1) t = fmaxf(t, __shfl_xor(t, o, 32));
            float mOld = mrw[qsub][st][r];
            float mNew = fmaxf(mOld, t);
            float al = __expf(mOld - mNew);
            float p0 = __expf(v0 - mNew);
            float p1 = __expf(v1 - mNew);
            float rs = p0 + p1;
            for (int o = 1; o < 16; o <<= 1) rs += __shfl_xor(rs, o, 32);
            mrw[qsub][st][r] = mNew;
            lrw[qsub][st][r] = lrw[qsub][st][r] * al + rs;
            for (int n = 0; n < 4; ++n) oacc[qsub][st][n][r] *= al;
            pscr[wave][combo][r + 8 * hf][l16] = (__bf16)p0;
            pscr[wave][combo][r + 8 * hf][16 + l16] = (__bf16)p1;
          }
        }
      asm volatile("s_wait_dscnt 0" ::: "memory");  // wave-local LDS RAW on pscr

      v16bf pa[2][2];
      for (int qsub = 0; qsub < 2; ++qsub)
        for (int st = 0; st < 2; ++st)
          for (int i = 0; i < 8; ++i) {
            int kk = kpat(i, hf);
            pa[qsub][st][2 * i] = pscr[wave][qsub * 2 + st][l16][kk];
            pa[qsub][st][2 * i + 1] = pscr[wave][qsub * 2 + st][l16][kk + 1];
          }
      for (int nt = 0; nt < 4; ++nt) {
        v16bf bv;
        for (int i = 0; i < 8; ++i) {
          int kk = kpat(i, hf);
          bv[2 * i] = vt[kb + kk][nt * 16 + l16];
          bv[2 * i + 1] = vt[kb + kk + 1][nt * 16 + l16];
        }
        for (int qsub = 0; qsub < 2; ++qsub)
          for (int st = 0; st < 2; ++st)
            oacc[qsub][st][nt] = __builtin_amdgcn_wmma_f32_16x16x32_bf16(
                false, pa[qsub][st], false, bv, (short)0, oacc[qsub][st][nt], false, false);
      }
    }
    __syncthreads();
  }

  float lam = 1.f / (1.f + __expf(-lambda_logit[h]));
  for (int qsub = 0; qsub < 2; ++qsub)
    for (int nt = 0; nt < 4; ++nt)
      for (int r = 0; r < 8; ++r) {
        float o1 = oacc[qsub][0][nt][r] / lrw[qsub][0][r];
        float o2 = oacc[qsub][1][nt][r] / lrw[qsub][1][r];
        int mrow = q_start + wave * 32 + qsub * 16 + r + 8 * hf;
        diffOut[((long long)b * KSEL + mrow) * DM + h * HD + nt * 16 + l16] = o1 - lam * o2;
      }
}

// ---------------- 9: RMS norm + scatter to output rows ----------------
__global__ __launch_bounds__(256) void rms_scatter_kernel(const float* __restrict__ proj,
                                                          const int* __restrict__ idx,
                                                          const float* __restrict__ rmsw,
                                                          float* __restrict__ out) {
  int wave = threadIdx.x >> 5, lane = threadIdx.x & 31;
  int row = blockIdx.x * 8 + wave;
  int b = row / KSEL, kr = row % KSEL;
  const float* pr = proj + ((long long)b * KSEL + kr) * DM;
  float ss = 0.f;
  for (int d = lane; d < DM; d += 32) { float v = pr[d]; ss += v * v; }
  for (int o = 1; o < 32; o <<= 1) ss += __shfl_xor(ss, o, 32);
  float sc = rsqrtf(ss * (1.f / DM) + RMSEPS);
  int orow = idx[b * KSEL + kr];
  float* op = out + ((long long)b * S_ + orow) * DM;
  for (int d = lane; d < DM; d += 32) op[d] = pr[d] * sc * rmsw[d];
}

// ---------------- launch ----------------
extern "C" void kernel_launch(void* const* d_in, const int* in_sizes, int n_in,
                              void* d_out, int out_size, void* d_ws, size_t ws_size,
                              hipStream_t stream) {
  const float* x     = (const float*)d_in[0];
  const float* cheby = (const float*)d_in[1];
  const float* Wqkv  = (const float*)d_in[2];
  const float* Wo    = (const float*)d_in[3];
  const float* rmsw  = (const float*)d_in[4];
  const float* lam   = (const float*)d_in[5];
  float* out = (float*)d_out;

  char* w = (char*)d_ws;
  float* scores = (float*)w;  w += (size_t)B_ * S_ * 4;
  int*   idx    = (int*)w;    w += (size_t)B_ * KSEL * 4;
  float* qkv    = (float*)w;  w += (size_t)B_ * KSEL * 4096 * 4;
  float* diff   = (float*)w;  w += (size_t)B_ * KSEL * DM * 4;
  float* proj   = (float*)w;  w += (size_t)B_ * KSEL * DM * 4;
  __bf16* xsel  = (__bf16*)w; w += (size_t)B_ * KSEL * DM * 2;
  __bf16* wqkvb = (__bf16*)w; w += (size_t)4 * DM * DM * 2;
  __bf16* wob   = (__bf16*)w; w += (size_t)DM * DM * 2;
  __bf16* diffb = (__bf16*)w; w += (size_t)B_ * KSEL * DM * 2;
  __bf16* qp    = (__bf16*)w; w += (size_t)B_ * NH * 2 * KSEL * HD * 2;
  __bf16* kp    = (__bf16*)w; w += (size_t)B_ * NH * KSEL * HD * 2;
  __bf16* vp    = (__bf16*)w; w += (size_t)B_ * NH * KSEL * HD * 2;

  int n4 = (B_ * S_ * DM) / 4;
  zero_kernel<<<(n4 + 255) / 256, 256, 0, stream>>>((float4*)out, n4);
  score_kernel<<<(B_ * (S_ - 1) + 7) / 8, 256, 0, stream>>>(x, cheby, scores);
  finalize_max_kernel<<<B_, 256, 0, stream>>>(scores);
  topk_kernel<<<B_, 1024, 0, stream>>>(scores, idx);
  cvt_bf16_kernel<<<(4 * DM * DM / 4 + 255) / 256, 256, 0, stream>>>(
      (const float4*)Wqkv, (ushort4*)wqkvb, 4 * DM * DM / 4);
  cvt_bf16_kernel<<<(DM * DM / 4 + 255) / 256, 256, 0, stream>>>(
      (const float4*)Wo, (ushort4*)wob, DM * DM / 4);
  gather_cvt_kernel<<<(B_ * KSEL * DM / 4) / 256, 256, 0, stream>>>(x, idx, (ushort4*)xsel);
  gemm_bf16<<<dim3(KSEL / 128, 4096 / 128, B_), 256, 0, stream>>>(
      xsel, (long long)KSEL * DM, DM, wqkvb, qkv, (long long)KSEL * 4096, 4096, 4096, DM);
  rope_pack_kernel<<<(B_ * KSEL * NH * 32) / 256, 256, 0, stream>>>(qkv, idx, qp, kp, vp);
  attn_kernel<<<dim3(KSEL / CHUNK_, NH, B_), 256, 0, stream>>>(qp, kp, vp, idx, lam, diff);
  cvt_bf16_kernel<<<(B_ * KSEL * DM / 4 + 255) / 256, 256, 0, stream>>>(
      (const float4*)diff, (ushort4*)diffb, B_ * KSEL * DM / 4);
  gemm_bf16<<<dim3(KSEL / 128, DM / 128, B_), 256, 0, stream>>>(
      diffb, (long long)KSEL * DM, DM, wob, proj, (long long)KSEL * DM, DM, DM, DM);
  rms_scatter_kernel<<<(B_ * KSEL) / 8, 256, 0, stream>>>(proj, idx, rmsw, out);
}
